// ImplicitWarpModule_46042049413501
// MI455X (gfx1250) — compile-verified, matrix operand order checked by reading, not run on hardware
//
#include <hip/hip_runtime.h>
#include <hip/hip_bf16.h>
#include <math.h>

typedef __attribute__((ext_vector_type(16))) _Float16 v16h;
typedef __attribute__((ext_vector_type(8)))  _Float16 v8h;
typedef __attribute__((ext_vector_type(8)))  float    v8f;

#define Nn    4
#define Cc    128
#define Hh    128
#define Ww    256
#define HWW   (Hh*Ww)            // 32768
#define DIMM  128
#define HD    16                 // head dim = 128/8
#define NWP_  4
#define TILE_P   64              // pixels per block
#define ROWS_FW  (TILE_P*NWP_)   // 256

// LDS bytes: FW 65536 + FC 16384 + K 65536 + V 65536 + Q 16384 + pb 2048 + gi 1024 + dec 512
#define SMEM_BYTES 232960

__device__ __forceinline__ float dimt(int i) {           // i in 0..63
    return __powf(10000.0f, (float)(2 * (i >> 1)) * (1.0f / 64.0f));
}
__device__ __forceinline__ float pe_elem(float base, int c) { // c in 0..127
    int i = c & 63;
    float arg = base / dimt(i);
    return (i & 1) ? __cosf(arg) : __sinf(arg);
}

__global__ __launch_bounds__(256)
void implicit_warp_attn(const float* __restrict__ feat_supp,
                        const float* __restrict__ feat_curr,
                        const float* __restrict__ flow,
                        const float* __restrict__ q_w, const float* __restrict__ q_b,
                        const float* __restrict__ k_w, const float* __restrict__ k_b,
                        const float* __restrict__ v_w, const float* __restrict__ v_b,
                        float* __restrict__ out)
{
    extern __shared__ char smem_raw[];
    _Float16* s_fw = (_Float16*)smem_raw;            // [256][128] feat_warp f16
    _Float16* s_fc = s_fw + ROWS_FW * DIMM;          // [64][128]  feat_curr f16
    _Float16* s_K  = s_fc + TILE_P * DIMM;           // [256][128] K
    _Float16* s_V  = s_K  + ROWS_FW * DIMM;          // [256][128] V
    _Float16* s_Q  = s_V  + ROWS_FW * DIMM;          // [64][128]  Q
    float* s_pb  = (float*)(s_Q + TILE_P * DIMM);    // [4][128] window pos bias
    int*   s_gi  = (int*)(s_pb + NWP_ * DIMM);       // [64][4] gather linear idx
    float* s_dec = (float*)(s_gi + TILE_P * NWP_);   // [64][2] fractional (y,x)

    const int tid    = threadIdx.x;
    const int lane   = tid & 31;
    const int wv     = tid >> 5;                     // 0..7 == head
    const int laneHi = lane >> 4;

    const int pix_base = blockIdx.x * TILE_P;
    const int nimg     = pix_base >> 15;             // / 32768
    const int pos_base = pix_base & (HWW - 1);
    const int y0       = pos_base >> 8;              // / 256 (tile stays in one row)
    const int x0       = pos_base & 255;

    const float TWO_PI = 6.283185307179586f;

    // ---------------- stage 0: pb table + per-pixel flow indices ----------------
    for (int t = tid; t < NWP_ * DIMM; t += 256) {
        int w4 = t >> 7, c = t & 127;
        float ny = (float)(w4 >> 1) * (TWO_PI / (1.0f + 1e-6f));
        float nx = (float)(w4 & 1)  * (TWO_PI / (1.0f + 1e-6f));
        s_pb[t] = pe_elem((c < 64) ? ny : nx, c);
    }
    if (tid < TILE_P) {
        int p = tid;
        int x = x0 + p;
        const float* fl = flow + (((size_t)nimg * Hh + y0) * Ww + x) * 2;
        float fx = fl[0], fy = fl[1];                 // flow.flip(-1): (dx,dy)->(dy,dx)
        float fully = (float)y0 + fy;
        float fullx = (float)x  + fx;
        float fy_f = floorf(fully), fx_f = floorf(fullx);
        s_dec[2 * p]     = fully - fy_f;
        s_dec[2 * p + 1] = fullx - fx_f;
        int giy = (int)fy_f, gix = (int)fx_f;
        #pragma unroll
        for (int w4 = 0; w4 < NWP_; ++w4) {
            int hy = giy + (w4 >> 1);
            int wx = gix + (w4 & 1);
            hy = (hy < 0) ? 0 : ((hy > Hh - 1) ? (Hh - 1) : hy);
            wx = (wx < 0) ? 0 : ((wx > Ww - 1) ? (Ww - 1) : wx);
            s_gi[p * NWP_ + w4] = hy * Ww + wx;       // duplicate-pad clamp
        }
    }
    __syncthreads();

    // ---------------- stage 1: gather + position encodings -> f16 in LDS -------
    #pragma unroll 1
    for (int t = tid; t < ROWS_FW * DIMM; t += 256) {
        int c = t >> 8;                                // channel (0..127)
        int r = t & 255;                               // row = p*4 + w (0..255)
        float v = feat_supp[((size_t)nimg * Cc + c) * HWW + s_gi[r]]
                + s_pb[(r & 3) * DIMM + c];
        s_fw[r * DIMM + c] = (_Float16)v;
    }
    #pragma unroll 1
    for (int t = tid; t < TILE_P * DIMM; t += 256) {
        int c = t >> 6;                                // channel (0..127)
        int p = t & 63;                                // pixel (0..63)
        float by = s_dec[2 * p]     * (TWO_PI / (2.0f + 1e-6f));
        float bx = s_dec[2 * p + 1] * (TWO_PI / (2.0f + 1e-6f));
        float v = feat_curr[((size_t)nimg * Cc + c) * HWW + pos_base + p]
                + pe_elem((c < 64) ? by : bx, c);
        s_fc[p * DIMM + c] = (_Float16)v;
    }
    __syncthreads();

    // ---------------- stage 2: WMMA projections (wave w owns cols 16w..16w+15) --
    const int ncol = wv * 16 + (lane & 15);

    // B fragments: 32x16 per k-step, WMMA 16-bit B layout built by transposing
    // W[k][n] (row-major, lanes 0-15 = consecutive n -> coalesced) with f32->f16.
    v16h bk[4], bv[4], bq[4];
    #pragma unroll
    for (int ks = 0; ks < 4; ++ks) {
        #pragma unroll
        for (int j = 0; j < 8; ++j) {
            int K0 = ks * 32 + ((j & 4) ? 16 : 0) + laneHi * 8 + 2 * (j & 3);
            bk[ks][2 * j]     = (_Float16)k_w[(size_t)K0 * DIMM + ncol];
            bk[ks][2 * j + 1] = (_Float16)k_w[(size_t)(K0 + 1) * DIMM + ncol];
            bv[ks][2 * j]     = (_Float16)v_w[(size_t)K0 * DIMM + ncol];
            bv[ks][2 * j + 1] = (_Float16)v_w[(size_t)(K0 + 1) * DIMM + ncol];
            bq[ks][2 * j]     = (_Float16)q_w[(size_t)K0 * DIMM + ncol];
            bq[ks][2 * j + 1] = (_Float16)q_w[(size_t)(K0 + 1) * DIMM + ncol];
        }
    }
    const float kbias = k_b[ncol], vbias = v_b[ncol], qbias = q_b[ncol];

    // K and V share the A fragment: 16 row-tiles x 4 k-steps x 2 wmma
    #pragma unroll 1
    for (int rt = 0; rt < ROWS_FW / 16; ++rt) {
        v8f accK = {}; v8f accV = {};
        #pragma unroll
        for (int ks = 0; ks < 4; ++ks) {
            // A 16x32 f16 fragment: two ds_load_b128 per lane (ISA 7.12.2 layout)
            const _Float16* ap = s_fw + (rt * 16 + (lane & 15)) * DIMM
                               + ks * 32 + laneHi * 8;
            v8h alo = *(const v8h*)ap;
            v8h ahi = *(const v8h*)(ap + 16);
            v16h a = __builtin_shufflevector(alo, ahi,
                        0,1,2,3,4,5,6,7,8,9,10,11,12,13,14,15);
            accK = __builtin_amdgcn_wmma_f32_16x16x32_f16(
                        false, a, false, bk[ks], (short)0, accK, false, false);
            accV = __builtin_amdgcn_wmma_f32_16x16x32_f16(
                        false, a, false, bv[ks], (short)0, accV, false, false);
        }
        #pragma unroll
        for (int g = 0; g < 8; ++g) {                 // C/D layout: row = g + 8*laneHi
            int row = rt * 16 + g + 8 * laneHi;
            s_K[row * DIMM + ncol] = (_Float16)(accK[g] + kbias);
            s_V[row * DIMM + ncol] = (_Float16)(accV[g] + vbias);
        }
    }
    // Q: 4 row-tiles (64 query rows)
    #pragma unroll 1
    for (int rt = 0; rt < TILE_P / 16; ++rt) {
        v8f accQ = {};
        #pragma unroll
        for (int ks = 0; ks < 4; ++ks) {
            const _Float16* ap = s_fc + (rt * 16 + (lane & 15)) * DIMM
                               + ks * 32 + laneHi * 8;
            v8h alo = *(const v8h*)ap;
            v8h ahi = *(const v8h*)(ap + 16);
            v16h a = __builtin_shufflevector(alo, ahi,
                        0,1,2,3,4,5,6,7,8,9,10,11,12,13,14,15);
            accQ = __builtin_amdgcn_wmma_f32_16x16x32_f16(
                        false, a, false, bq[ks], (short)0, accQ, false, false);
        }
        #pragma unroll
        for (int g = 0; g < 8; ++g) {
            int row = rt * 16 + g + 8 * laneHi;
            s_Q[row * DIMM + ncol] = (_Float16)(accQ[g] + qbias);
        }
    }
    __syncthreads();

    // ---------------- stage 3: per-(pixel, head) softmax attention --------------
    // 64 pixels x 8 heads = 512 items, 2 per thread; vector (b128) LDS reads.
    #pragma unroll
    for (int it = tid; it < TILE_P * 8; it += 256) {
        int p  = it >> 3;                              // 0..63
        int h  = it & 7;                               // 0..7
        int cb = h * HD;
        const v8h* qv = (const v8h*)(s_Q + p * DIMM + cb);
        v8h q0 = qv[0], q1 = qv[1];
        float logit[4];
        #pragma unroll
        for (int w4 = 0; w4 < 4; ++w4) {
            const v8h* kv = (const v8h*)(s_K + (p * 4 + w4) * DIMM + cb);
            v8h k0 = kv[0], k1 = kv[1];
            float s = 0.f;
            #pragma unroll
            for (int d = 0; d < 8; ++d)
                s += (float)q0[d] * (float)k0[d] + (float)q1[d] * (float)k1[d];
            logit[w4] = s * 0.25f;                     // hd^-0.5 = 16^-0.5
        }
        float m = fmaxf(fmaxf(logit[0], logit[1]), fmaxf(logit[2], logit[3]));
        float e[4]; float sum = 0.f;
        #pragma unroll
        for (int w4 = 0; w4 < 4; ++w4) { e[w4] = __expf(logit[w4] - m); sum += e[w4]; }
        float inv = 1.0f / sum;
        float o[HD];
        #pragma unroll
        for (int d = 0; d < HD; ++d) o[d] = 0.f;
        #pragma unroll
        for (int w4 = 0; w4 < 4; ++w4) {
            const v8h* vv = (const v8h*)(s_V + (p * 4 + w4) * DIMM + cb);
            v8h v0 = vv[0], v1 = vv[1];
            #pragma unroll
            for (int d = 0; d < 8; ++d) {
                o[d]     += e[w4] * (float)v0[d];
                o[d + 8] += e[w4] * (float)v1[d];
            }
        }
        float* op = out + ((size_t)nimg * DIMM + cb) * HWW + pos_base + p;
        #pragma unroll
        for (int d = 0; d < HD; ++d)
            op[(size_t)d * HWW] = o[d] * inv;          // out is (N, DIM, H, W)
    }
}

extern "C" void kernel_launch(void* const* d_in, const int* in_sizes, int n_in,
                              void* d_out, int out_size, void* d_ws, size_t ws_size,
                              hipStream_t stream) {
    const float* feat_supp = (const float*)d_in[0];
    const float* feat_curr = (const float*)d_in[1];
    const float* flow      = (const float*)d_in[2];
    const float* q_w       = (const float*)d_in[3];
    const float* q_b       = (const float*)d_in[4];
    const float* k_w       = (const float*)d_in[5];
    const float* k_b       = (const float*)d_in[6];
    const float* v_w       = (const float*)d_in[7];
    const float* v_b       = (const float*)d_in[8];
    float* out             = (float*)d_out;

    (void)in_sizes; (void)n_in; (void)out_size; (void)d_ws; (void)ws_size;

    hipFuncSetAttribute(reinterpret_cast<const void*>(implicit_warp_attn),
                        hipFuncAttributeMaxDynamicSharedMemorySize, SMEM_BYTES);

    const int blocks = (Nn * HWW) / TILE_P;            // 2048
    implicit_warp_attn<<<blocks, 256, SMEM_BYTES, stream>>>(
        feat_supp, feat_curr, flow, q_w, q_b, k_w, k_b, v_w, v_b, out);
}
